// StackGCNEncoder_75093208203379
// MI455X (gfx1250) — matrix operand order repaired and checked by reading, not compiled, not run on hardware
//
#include <hip/hip_runtime.h>
#include <hip/hip_bf16.h>

// ---------------------------------------------------------------------------
// StackGCNEncoder for MI455X (gfx1250, wave32, WMMA + Tensor Data Mover)
//
// Dense transforms: bf16 WMMA GEMMs (v_wmma_f32_16x16x32_bf16, f32 accum).
// A-tiles are staged HBM->LDS by the TDM (tensor_load_to_lds) with
// DOUBLE-BUFFERED descriptors: wave0 issues tile i+1 while the workgroup
// computes on tile i, waiting s_wait_tensorcnt 1 (per-wave in-order TDM
// completion).  D# pad fields give the bank-conflict-free 40-element LDS
// pitch; tensor_dim OOB rules zero-fill partial M blocks.
// All 4 relations are folded into the GEMM N dimension, so tmp comes out in
// [n][s*k] layout == agg layout; the sparse aggregation is then symmetric
// L2-resident f32 atomics on top of the self-loop GEMM output, and a fused
// relu epilogue emits next-layer bf16 activations / final f32 output.
// ---------------------------------------------------------------------------

typedef __bf16 bf16;
typedef __attribute__((ext_vector_type(8)))  __bf16 v8bf;
typedef __attribute__((ext_vector_type(16))) __bf16 v16bf;
typedef __attribute__((ext_vector_type(8)))  float  v8f;
typedef __attribute__((ext_vector_type(4)))  unsigned int u32x4;
typedef __attribute__((ext_vector_type(4)))  int i32x4;
typedef __attribute__((ext_vector_type(8)))  int i32x8;

#define BM   128
#define BLKN 128
#define BK   32
#define LDA  40   // LDS row pitch (elements): 80B rows -> 16B aligned, conflict-free

#if __has_builtin(__builtin_amdgcn_tensor_load_to_lds)
#define USE_TDM 1
#else
#define USE_TDM 0
#endif

#if USE_TDM
// One TDM descriptor moving a [min(remRows,128) x 32] bf16 tile into LDS at
// ldsOff; LDS row pitch = 40 elements via pad fields (16 DW data + 4 DW pad).
__device__ __forceinline__ void tdm_load_tile_a(const bf16* gsrc, unsigned ldsOff,
                                                unsigned remRows, unsigned remCols,
                                                unsigned rowStrideElems)
{
  unsigned long long ga = (unsigned long long)(uintptr_t)gsrc;
  u32x4 g0;
  g0[0] = 1u;                                            // count=1 (valid user D#)
  g0[1] = ldsOff;                                        // lds_addr
  g0[2] = (unsigned)(ga & 0xffffffffu);                  // global_addr[31:0]
  g0[3] = (unsigned)((ga >> 32) & 0x01ffffffu)           // global_addr[56:32]
        | (2u << 30);                                    // type=2 ("image")

  i32x8 g1;
  // data_size=1 (2B), pad_enable=1, pad_interval=3 (16 DW), pad_amount=3 (4 DW)
  g1[0] = (1 << 16) | (1 << 20) | (3 << 22) | (3 << 25);
  g1[1] = (int)((remCols & 0xffffu) << 16);              // tensor_dim0[15:0]
  g1[2] = (int)((remCols >> 16) | ((remRows & 0xffffu) << 16)); // dim0 hi / dim1 lo
  g1[3] = (int)((remRows >> 16) | ((unsigned)BK << 16)); // dim1 hi / tile_dim0=32
  g1[4] = (int)BM;                                       // tile_dim1=128, tile_dim2=0
  g1[5] = (int)rowStrideElems;                           // tensor_dim0_stride lo32
  g1[6] = 0;
  g1[7] = 0;
  i32x4 gz = {0, 0, 0, 0};
#if defined(__clang_major__) && __clang_major__ >= 23
  i32x8 gz8 = {0, 0, 0, 0, 0, 0, 0, 0};
  __builtin_amdgcn_tensor_load_to_lds(g0, g1, gz, gz, gz8, 0);
#else
  __builtin_amdgcn_tensor_load_to_lds(g0, g1, gz, gz, 0);
#endif
}
#endif

// C = A @ B ; A: [M,K] bf16 row-major, BT: [N,K] bf16 row-major (B transposed),
// C: [M,N] f32.  N compile-time -> immediate store offsets.
// Block tile 128x128, 8 waves as 4(M) x 2(N); wave tile 32x64 (2x4 WMMA frags).
template <int N>
__global__ __launch_bounds__(256) void gemm_bf16_wmma(
    const bf16* __restrict__ A,
    const bf16* __restrict__ BT,
    float* __restrict__ C,
    int M, int K)
{
  __shared__ bf16 Atile[2][BM * LDA];

  const int tid  = threadIdx.x;
  const int lane = tid & 31;
  const int wave = tid >> 5;
  const int wm   = wave >> 1;      // 0..3
  const int wn   = wave & 1;       // 0..1
  const int half = lane >> 4;      // K-half selector per WMMA bf16 layout
  const int l15  = lane & 15;

  const int mBlock = blockIdx.x * BM;
  const int nBlock = blockIdx.y * BLKN;

  v8f acc[2][4] = {};

  const int iters = K / BK;

#if USE_TDM
  const unsigned aOff0 = (unsigned)(uintptr_t)(&Atile[0][0]);
  const unsigned aOff1 = (unsigned)(uintptr_t)(&Atile[1][0]);
  if (wave == 0) {
    tdm_load_tile_a(A + (long long)mBlock * K, aOff0,
                    (unsigned)(M - mBlock), (unsigned)K, (unsigned)K);
  }
#else
  const int crow = tid >> 1;
  const int ccol = (tid & 1) * 16;
  int grow = mBlock + crow;
  if (grow >= M) grow = M - 1;
#endif

  for (int i = 0; i < iters; ++i) {
    const int k0 = i * BK;
#if USE_TDM
    if (wave == 0) {
      if (i + 1 < iters) {
        // prefetch next tile into the other buffer, then wait for current
        tdm_load_tile_a(A + (long long)mBlock * K + (k0 + BK),
                        ((i + 1) & 1) ? aOff1 : aOff0,
                        (unsigned)(M - mBlock), (unsigned)(K - k0 - BK),
                        (unsigned)K);
        __builtin_amdgcn_s_wait_tensorcnt(1);
      } else {
        __builtin_amdgcn_s_wait_tensorcnt(0);
      }
    }
    __syncthreads();                       // publish tile i
    const bf16* At = &Atile[i & 1][0];
#else
    __syncthreads();
    v16bf av = *(const v16bf*)(A + (long long)grow * K + k0 + ccol);
    *(v16bf*)(&Atile[0][crow * LDA + ccol]) = av;
    __syncthreads();
    const bf16* At = &Atile[0][0];
#endif

    v16bf afrag[2], bfrag[4];
    #pragma unroll
    for (int mi = 0; mi < 2; ++mi) {
      // 16x32 bf16 A frag: lanes 0-15 K {0..7,16..23}, lanes 16-31 K {8..15,24..31}
      int mrow = wm * 32 + mi * 16 + l15;
      union { v16bf v; v8bf h[2]; } u;
      u.h[0] = *(const v8bf*)(At + mrow * LDA + half * 8);
      u.h[1] = *(const v8bf*)(At + mrow * LDA + half * 8 + 16);
      afrag[mi] = u.v;
    }
    #pragma unroll
    for (int ni = 0; ni < 4; ++ni) {
      int ncol = nBlock + wn * 64 + ni * 16 + l15;
      union { v16bf v; v8bf h[2]; } ub;
      ub.h[0] = *(const v8bf*)(BT + (long long)ncol * K + k0 + half * 8);
      ub.h[1] = *(const v8bf*)(BT + (long long)ncol * K + k0 + half * 8 + 16);
      bfrag[ni] = ub.v;
    }
    #pragma unroll
    for (int mi = 0; mi < 2; ++mi)
      #pragma unroll
      for (int ni = 0; ni < 4; ++ni)
        acc[mi][ni] = __builtin_amdgcn_wmma_f32_16x16x32_bf16(
            false, afrag[mi], false, bfrag[ni], (short)0, acc[mi][ni],
            false, false);

#if USE_TDM
    __syncthreads();   // all reads of this buffer done before wave0 re-targets it
#endif
  }

  // C layout: VGPR r -> M = base + half*8 + r, N = base + (lane&15)
  const bool fullBlock = (mBlock + BM) <= M;
  #pragma unroll
  for (int mi = 0; mi < 2; ++mi) {
    #pragma unroll
    for (int ni = 0; ni < 4; ++ni) {
      const int n  = nBlock + wn * 64 + ni * 16 + l15;
      const int m0 = mBlock + wm * 32 + mi * 16 + half * 8;
      float* cp = C + (long long)m0 * N + n;
      if (fullBlock) {
        #pragma unroll
        for (int r = 0; r < 8; ++r)
          cp[r * N] = acc[mi][ni][r];     // immediate offsets (N constexpr)
      } else {
        #pragma unroll
        for (int r = 0; r < 8; ++r)
          if (m0 + r < M) cp[r * N] = acc[mi][ni][r];
      }
    }
  }
}

// f32 -> bf16, 4 elements/thread
__global__ void to_bf16_kernel(const float* __restrict__ in, bf16* __restrict__ out,
                               long long n4)
{
  long long i = ((long long)blockIdx.x * blockDim.x + threadIdx.x);
  if (i >= n4) return;
  float4 f = *(const float4*)(in + 4 * i);
  bf16* o = out + 4 * i;
  o[0] = (bf16)f.x; o[1] = (bf16)f.y; o[2] = (bf16)f.z; o[3] = (bf16)f.w;
}

// Weight transpose+convert: in [z][D,Nc] f32 row-major -> out [(z*Nc)+n][D] bf16
__global__ void transpose_bf16_kernel(const float* __restrict__ in, bf16* __restrict__ out,
                                      int D, int Nc)
{
  long long base = (long long)blockIdx.z * D * Nc;
  long long i = (long long)blockIdx.x * blockDim.x + threadIdx.x;
  if (i >= (long long)D * Nc) return;
  int d = (int)(i / Nc);
  int n = (int)(i - (long long)d * Nc);
  out[base + (long long)n * D + d] = (bf16)in[base + i];
}

// Graph aggregation: agg[row, s*k + j..j+3] += val * tmp[col, s*k + j..j+3]
// (tmp and agg share the [node][S*k] row layout)
__global__ void spmm_scatter_kernel(const int* __restrict__ rows,
                                    const int* __restrict__ cols,
                                    const float* __restrict__ vals,
                                    const float* __restrict__ tmp,   // [nSrc][Sk]
                                    float* __restrict__ agg,         // [nDst][Sk]
                                    int E, int kLog2, int Sk)
{
  const int s = blockIdx.y;
  long long t = (long long)blockIdx.x * blockDim.x + threadIdx.x;
  if (t >= ((long long)E << (kLog2 - 2))) return;
  int e  = (int)(t >> (kLog2 - 2));
  int j4 = (int)(t & ((1 << (kLog2 - 2)) - 1)) << 2;
  const int*   rs = rows + (long long)s * E;
  const int*   cs = cols + (long long)s * E;
  const float* vs = vals + (long long)s * E;
  int   r = rs[e];
  int   c = cs[e];
  float v = vs[e];
  int   off = (s << kLog2) + j4;
  float4 f = *(const float4*)(tmp + (long long)c * Sk + off);
  float* dst = agg + (long long)r * Sk + off;
  atomicAdd(dst + 0, v * f.x);
  atomicAdd(dst + 1, v * f.y);
  atomicAdd(dst + 2, v * f.z);
  atomicAdd(dst + 3, v * f.w);
}

// relu epilogue, 4 elements/thread; optional f32 (final) and bf16 (next layer)
__global__ void relu_emit_kernel(const float* __restrict__ agg,
                                 float* __restrict__ out32,
                                 bf16*  __restrict__ out16,
                                 long long n4)
{
  long long i = (long long)blockIdx.x * blockDim.x + threadIdx.x;
  if (i >= n4) return;
  float4 f = *(const float4*)(agg + 4 * i);
  f.x = f.x > 0.f ? f.x : 0.f;
  f.y = f.y > 0.f ? f.y : 0.f;
  f.z = f.z > 0.f ? f.z : 0.f;
  f.w = f.w > 0.f ? f.w : 0.f;
  if (out32) *(float4*)(out32 + 4 * i) = f;
  if (out16) {
    bf16* o = out16 + 4 * i;
    o[0] = (bf16)f.x; o[1] = (bf16)f.y; o[2] = (bf16)f.z; o[3] = (bf16)f.w;
  }
}

static inline unsigned ceil_div_u(long long a, long long b) {
  return (unsigned)((a + b - 1) / b);
}

extern "C" void kernel_launch(void* const* d_in, const int* in_sizes, int n_in,
                              void* d_out, int out_size, void* d_ws, size_t ws_size,
                              hipStream_t stream) {
  const int S = 4, E = 500000, NR = 50000, NP = 20000;

  const int*   rna_rows  = (const int*)  d_in[0];
  const int*   rna_cols  = (const int*)  d_in[1];
  const float* rna_vals  = (const float*)d_in[2];
  const int*   prot_rows = (const int*)  d_in[3];
  const int*   prot_cols = (const int*)  d_in[4];
  const float* prot_vals = (const float*)d_in[5];
  const float* H_rna     = (const float*)d_in[6];   // [50000,1024]
  const float* H_prot    = (const float*)d_in[7];   // [20000,1024]
  const float* w0        = (const float*)d_in[8];   // [4,1024,128]
  const float* sw0       = (const float*)d_in[9];   // [1024,512]
  const float* w1        = (const float*)d_in[10];  // [4,512,64]
  const float* sw1       = (const float*)d_in[11];  // [512,256]

  // ---- workspace bump allocator (256B aligned) ----
  char* wp = (char*)d_ws;
  auto balloc = [&](size_t bytes) -> char* {
    char* p = wp;
    wp += (bytes + 255) & ~(size_t)255;
    return p;
  };
  bf16*  hr16  = (bf16*) balloc((size_t)NR * 1024 * sizeof(bf16));
  bf16*  hp16  = (bf16*) balloc((size_t)NP * 1024 * sizeof(bf16));
  bf16*  wt16  = (bf16*) balloc((size_t)S * 1024 * 128 * sizeof(bf16)); // [S*Kc][D]
  bf16*  swt16 = (bf16*) balloc((size_t)1024 * 512 * sizeof(bf16));     // [Sk][D]
  float* tmpu  = (float*)balloc((size_t)NR * 512 * sizeof(float));      // [NR][S*Kc]
  float* tmpv  = (float*)balloc((size_t)NP * 512 * sizeof(float));      // [NP][S*Kc]
  float* aggr  = (float*)balloc((size_t)NR * 512 * sizeof(float));      // [NR][Sk]
  float* aggp  = (float*)balloc((size_t)NP * 512 * sizeof(float));      // [NP][Sk]

  // ---- initial activation conversion to bf16 ----
  to_bf16_kernel<<<ceil_div_u((long long)NR * 1024 / 4, 256), 256, 0, stream>>>(
      H_rna, hr16, (long long)NR * 1024 / 4);
  to_bf16_kernel<<<ceil_div_u((long long)NP * 1024 / 4, 256), 256, 0, stream>>>(
      H_prot, hp16, (long long)NP * 1024 / 4);

  struct LayerCfg { int D, Kc, kLog2, Sk; const float *W, *SW; };
  const LayerCfg layers[2] = {
      {1024, 128, 7, 512, w0, sw0},
      { 512,  64, 6, 256, w1, sw1},
  };

  for (int l = 0; l < 2; ++l) {
    const LayerCfg& cfg = layers[l];
    const int D = cfg.D, Kc = cfg.Kc, Sk = cfg.Sk;

    // 1) convert+transpose weights: W [S][D,Kc] -> wt16 [S*Kc][D]; SW [D,Sk] -> [Sk][D]
    {
      dim3 g(ceil_div_u((long long)D * Kc, 256), 1, S);
      transpose_bf16_kernel<<<g, 256, 0, stream>>>(cfg.W, wt16, D, Kc);
      dim3 g2(ceil_div_u((long long)D * Sk, 256), 1, 1);
      transpose_bf16_kernel<<<g2, 256, 0, stream>>>(cfg.SW, swt16, D, Sk);
    }

    // 2) relation transforms folded into N: tmp = H @ [W_0|...|W_3]  ([n][S*Kc])
    // 3) self-loop GEMM seeds agg: agg = H @ SW                      ([n][Sk])
    if (l == 0) {
      gemm_bf16_wmma<512><<<dim3(ceil_div_u(NR, BM), 512 / BLKN), 256, 0, stream>>>(
          hr16, wt16, tmpu, NR, D);
      gemm_bf16_wmma<512><<<dim3(ceil_div_u(NP, BM), 512 / BLKN), 256, 0, stream>>>(
          hp16, wt16, tmpv, NP, D);
      gemm_bf16_wmma<512><<<dim3(ceil_div_u(NR, BM), 512 / BLKN), 256, 0, stream>>>(
          hr16, swt16, aggr, NR, D);
      gemm_bf16_wmma<512><<<dim3(ceil_div_u(NP, BM), 512 / BLKN), 256, 0, stream>>>(
          hp16, swt16, aggp, NP, D);
    } else {
      gemm_bf16_wmma<256><<<dim3(ceil_div_u(NR, BM), 256 / BLKN), 256, 0, stream>>>(
          hr16, wt16, tmpu, NR, D);
      gemm_bf16_wmma<256><<<dim3(ceil_div_u(NP, BM), 256 / BLKN), 256, 0, stream>>>(
          hp16, wt16, tmpv, NP, D);
      gemm_bf16_wmma<256><<<dim3(ceil_div_u(NR, BM), 256 / BLKN), 256, 0, stream>>>(
          hr16, swt16, aggr, NR, D);
      gemm_bf16_wmma<256><<<dim3(ceil_div_u(NP, BM), 256 / BLKN), 256, 0, stream>>>(
          hp16, swt16, aggp, NP, D);
    }

    // 4) sparse graph aggregation (L2-resident f32 atomics), 4 features/thread
    {
      dim3 ga(ceil_div_u((long long)E << (cfg.kLog2 - 2), 256), S, 1);
      spmm_scatter_kernel<<<ga, 256, 0, stream>>>(rna_rows, rna_cols, rna_vals,
                                                  tmpv, aggr, E, cfg.kLog2, Sk);
      spmm_scatter_kernel<<<ga, 256, 0, stream>>>(prot_rows, prot_cols, prot_vals,
                                                  tmpu, aggp, E, cfg.kLog2, Sk);
    }

    // 5) relu epilogue: next-layer bf16 activations, or final f32 output
    if (l == 0) {
      relu_emit_kernel<<<ceil_div_u((long long)NR * Sk / 4, 256), 256, 0, stream>>>(
          aggr, nullptr, hr16, (long long)NR * Sk / 4);
      relu_emit_kernel<<<ceil_div_u((long long)NP * Sk / 4, 256), 256, 0, stream>>>(
          aggp, nullptr, hp16, (long long)NP * Sk / 4);
    } else {
      float* out = (float*)d_out;  // concat(H_rna [NR,256], H_prot [NP,256])
      relu_emit_kernel<<<ceil_div_u((long long)NR * Sk / 4, 256), 256, 0, stream>>>(
          aggr, out, nullptr, (long long)NR * Sk / 4);
      relu_emit_kernel<<<ceil_div_u((long long)NP * Sk / 4, 256), 256, 0, stream>>>(
          aggp, out + (long long)NR * Sk, nullptr, (long long)NP * Sk / 4);
    }
  }
  (void)in_sizes; (void)n_in; (void)out_size; (void)ws_size;
}
